// DeformableAlignment_72748156059678
// MI455X (gfx1250) — matrix operand order repaired and checked by reading, not compiled, not run on hardware
//
#include <hip/hip_runtime.h>
#include <hip/hip_bf16.h>

typedef __attribute__((ext_vector_type(16))) _Float16 v16h;
typedef __attribute__((ext_vector_type(8)))  float    v8f;

#define NB 8
#define NC 64
#define NH 128
#define NW 128

// ---------------------------------------------------------------------------
// Weight packing: lay out A-fragments (16x32 f16 per lane-set) so that each
// lane of a wave loads its 16 halves with one contiguous 32B read.
// Packed index = ((kb*numM + m)*32 + lane)*16 + e
// A-fragment element mapping (contiguous k order, same as the LDS B reads):
//   row    = lane & 15
//   Klocal = e + (lane>=16 ? 16 : 0)
// K (global) = kb*32 + Klocal;  K = tap*cinDim + cin  (tap-major)
// ---------------------------------------------------------------------------
__global__ void pack_weights(const float* __restrict__ w, _Float16* __restrict__ wp,
                             int numM, int cinDim, int outCh, int total) {
  int idx = blockIdx.x * blockDim.x + threadIdx.x;
  if (idx >= total) return;
  int e    = idx & 15;
  int lane = (idx >> 4) & 31;
  int km   = idx >> 9;
  int m    = km % numM;
  int kb   = km / numM;
  int row  = lane & 15;
  int klocal = e + ((lane >= 16) ? 16 : 0);
  int K    = kb * 32 + klocal;
  int tap  = K / cinDim;
  int cin  = K - tap * cinDim;
  int och  = m * 16 + row;
  float v = 0.0f;
  if (och < outCh)
    v = w[((size_t)(och * cinDim + cin) * 3 + tap / 3) * 3 + (tap % 3)];
  wp[idx] = (_Float16)v;
}

// ---------------------------------------------------------------------------
// Conv1: 128->64 ch, 3x3 SAME, bias + leaky ReLU(0.1), f16 output (hmid).
// Implicit GEMM: M=64 (4 m-tiles), N=64 pixels/WG (4 n-tiles), K=9*128=1152
// (36 K-blocks of 32, tap-major so one (dy,dx) per K-block).
// Block = 128 threads (4 waves); wave w owns m-tile w across all 4 n-tiles.
// ---------------------------------------------------------------------------
__global__ __launch_bounds__(128)
void conv1_wmma(const float* __restrict__ ref, const float* __restrict__ tgt,
                const _Float16* __restrict__ w1p, const float* __restrict__ b1,
                _Float16* __restrict__ hmid) {
  __shared__ _Float16 Bl[32 * 64];            // B-tile: rows = pixel, cols = Klocal
  const int b    = blockIdx.z;
  const int y    = blockIdx.y;
  const int x0   = blockIdx.x * 64;
  const int t    = threadIdx.x;
  const int wave = t >> 5;
  const int lane = t & 31;

  v8f acc0 = {}, acc1 = {}, acc2 = {}, acc3 = {};

  const int krow = t >> 2;                    // 0..31 : K row staged by this thread
  const int col0 = (t & 3) * 16;              // 16-col chunk

  for (int kb = 0; kb < 36; ++kb) {
    const int tap = kb >> 2;
    const int dy  = tap / 3 - 1, dx = tap % 3 - 1;
    const int cin = (kb & 3) * 32 + krow;
    const int yy  = y + dy;
    const int yc  = yy < 0 ? 0 : (yy > NH - 1 ? NH - 1 : yy);
    const bool yok = (yy >= 0) && (yy < NH);
    const float* src = (cin < NC)
        ? (ref + (((size_t)b * NC + cin)       * NH + yc) * NW)
        : (tgt + (((size_t)b * NC + (cin - NC)) * NH + yc) * NW);
#pragma unroll
    for (int j = 0; j < 16; ++j) {
      int xx = x0 + col0 + j + dx;
      float v = (yok && xx >= 0 && xx < NW) ? src[xx] : 0.0f;
      Bl[(col0 + j) * 32 + krow] = (_Float16)v;
    }
    __syncthreads();

    v16h a = *(const v16h*)(w1p + ((size_t)(kb * 4 + wave) * 32 + lane) * 16);
    v16h f0 = *(const v16h*)(Bl + ( 0 + (lane & 15)) * 32 + (lane >> 4) * 16);
    v16h f1 = *(const v16h*)(Bl + (16 + (lane & 15)) * 32 + (lane >> 4) * 16);
    v16h f2 = *(const v16h*)(Bl + (32 + (lane & 15)) * 32 + (lane >> 4) * 16);
    v16h f3 = *(const v16h*)(Bl + (48 + (lane & 15)) * 32 + (lane >> 4) * 16);
    acc0 = __builtin_amdgcn_wmma_f32_16x16x32_f16(false, a, false, f0, (short)0, acc0, false, false);
    acc1 = __builtin_amdgcn_wmma_f32_16x16x32_f16(false, a, false, f1, (short)0, acc1, false, false);
    acc2 = __builtin_amdgcn_wmma_f32_16x16x32_f16(false, a, false, f2, (short)0, acc2, false, false);
    acc3 = __builtin_amdgcn_wmma_f32_16x16x32_f16(false, a, false, f3, (short)0, acc3, false, false);
    __syncthreads();
  }

  // C/D layout: lane<16 -> M=r, N=lane; lane>=16 -> M=r+8, N=lane-16
  const int hi8 = (lane >> 4) * 8;
  const int n   = lane & 15;
#pragma unroll
  for (int r = 0; r < 8; ++r) {
    const int och = wave * 16 + hi8 + r;
    const float bv = b1[och];
    const size_t base = (((size_t)b * NC + och) * NH + y) * NW + x0 + n;
    float v;
    v = acc0[r] + bv; v = (v >= 0.f) ? v : 0.1f * v; hmid[base +  0] = (_Float16)v;
    v = acc1[r] + bv; v = (v >= 0.f) ? v : 0.1f * v; hmid[base + 16] = (_Float16)v;
    v = acc2[r] + bv; v = (v >= 0.f) ? v : 0.1f * v; hmid[base + 32] = (_Float16)v;
    v = acc3[r] + bv; v = (v >= 0.f) ? v : 0.1f * v; hmid[base + 48] = (_Float16)v;
  }
}

// ---------------------------------------------------------------------------
// Conv2: 64->18 ch, 3x3 SAME + bias -> offset map (f32).
// M padded to 32 (2 m-tiles), N=64 pixels, K=9*64=576 (18 K-blocks).
// 4 waves: wave = (m, ngroup); each wave does 2 n-tiles.
// ---------------------------------------------------------------------------
__global__ __launch_bounds__(128)
void conv2_wmma(const _Float16* __restrict__ hmid, const _Float16* __restrict__ w2p,
                const float* __restrict__ b2, float* __restrict__ offs) {
  __shared__ _Float16 Bl[32 * 64];
  const int b    = blockIdx.z;
  const int y    = blockIdx.y;
  const int x0   = blockIdx.x * 64;
  const int t    = threadIdx.x;
  const int wave = t >> 5;
  const int lane = t & 31;
  const int m    = wave & 1;
  const int ng   = wave >> 1;

  v8f acc0 = {}, acc1 = {};
  const int krow = t >> 2;
  const int col0 = (t & 3) * 16;

  for (int kb = 0; kb < 18; ++kb) {
    const int tap = kb >> 1;
    const int dy  = tap / 3 - 1, dx = tap % 3 - 1;
    const int cin = (kb & 1) * 32 + krow;
    const int yy  = y + dy;
    const int yc  = yy < 0 ? 0 : (yy > NH - 1 ? NH - 1 : yy);
    const bool yok = (yy >= 0) && (yy < NH);
    const _Float16* src = hmid + (((size_t)b * NC + cin) * NH + yc) * NW;
#pragma unroll
    for (int j = 0; j < 16; ++j) {
      int xx = x0 + col0 + j + dx;
      _Float16 v = (yok && xx >= 0 && xx < NW) ? src[xx] : (_Float16)0.0f;
      Bl[(col0 + j) * 32 + krow] = v;
    }
    __syncthreads();

    v16h a  = *(const v16h*)(w2p + ((size_t)(kb * 2 + m) * 32 + lane) * 16);
    v16h f0 = *(const v16h*)(Bl + (ng * 32 + (lane & 15)) * 32 + (lane >> 4) * 16);
    v16h f1 = *(const v16h*)(Bl + (ng * 32 + 16 + (lane & 15)) * 32 + (lane >> 4) * 16);
    acc0 = __builtin_amdgcn_wmma_f32_16x16x32_f16(false, a, false, f0, (short)0, acc0, false, false);
    acc1 = __builtin_amdgcn_wmma_f32_16x16x32_f16(false, a, false, f1, (short)0, acc1, false, false);
    __syncthreads();
  }

  const int hi8 = (lane >> 4) * 8;
  const int n   = lane & 15;
#pragma unroll
  for (int r = 0; r < 8; ++r) {
    const int och = m * 16 + hi8 + r;
    if (och < 18) {
      const float bv = b2[och];
      const size_t base = (((size_t)b * 18 + och) * NH + y) * NW + x0 + ng * 32 + n;
      offs[base +  0] = acc0[r] + bv;
      offs[base + 16] = acc1[r] + bv;
    }
  }
}

// ---------------------------------------------------------------------------
// Deformable conv: bilinear-gather target_feat per (pixel, tap, cin), stage
// 32x16 f16 B-tiles, WMMA against packed wd. M=64 (4 m-tiles = 4 waves),
// N=16 pixels/WG, K=9*64=576 (18 K-blocks, tap-major).
// Sample parameters (y0,x0,wy,wx) computed once per WG from the offset map.
// ---------------------------------------------------------------------------
__device__ __forceinline__ float dsample(const float* __restrict__ p, int yi, int xi) {
  if (yi < 0 || yi >= NH || xi < 0 || xi >= NW) return 0.0f;
  return p[yi * NW + xi];
}

__global__ __launch_bounds__(128)
void deform_wmma(const float* __restrict__ tgt, const float* __restrict__ offs,
                 const _Float16* __restrict__ wdp, float* __restrict__ out) {
  __shared__ _Float16 Bl[32 * 16];
  __shared__ int   sy0[144], sx0[144];
  __shared__ float swy[144], swx[144];

  const int b    = blockIdx.z;
  const int y    = blockIdx.y;
  const int x0   = blockIdx.x * 16;
  const int t    = threadIdx.x;
  const int wave = t >> 5;
  const int lane = t & 31;

  // Precompute 9 taps x 16 pixels of bilinear parameters.
  for (int i = t; i < 144; i += 128) {
    const int tap = i / 16, n = i % 16;
    const int x = x0 + n;
    const float dyv = offs[(((size_t)b * 18 + tap * 2 + 0) * NH + y) * NW + x];
    const float dxv = offs[(((size_t)b * 18 + tap * 2 + 1) * NH + y) * NW + x];
    const float ys = (float)y + (float)(tap / 3 - 1) + dyv;
    const float xs = (float)x + (float)(tap % 3 - 1) + dxv;
    const float yf = floorf(ys), xf = floorf(xs);
    sy0[i] = (int)yf;  sx0[i] = (int)xf;
    swy[i] = ys - yf;  swx[i] = xs - xf;
  }
  __syncthreads();

  v8f acc = {};
  const int krow = t >> 2;          // 0..31
  const int nb   = (t & 3) * 4;     // 4 pixels per thread

  for (int kb = 0; kb < 18; ++kb) {
    const int tap = kb >> 1;
    const int cin = (kb & 1) * 32 + krow;
    const float* plane = tgt + ((size_t)b * NC + cin) * (size_t)(NH * NW);
#pragma unroll
    for (int j = 0; j < 4; ++j) {
      const int n  = nb + j;
      const int id = tap * 16 + n;
      const int yi = sy0[id], xi = sx0[id];
      const float wy = swy[id], wx = swx[id];
      const float v00 = dsample(plane, yi,     xi);
      const float v01 = dsample(plane, yi,     xi + 1);
      const float v10 = dsample(plane, yi + 1, xi);
      const float v11 = dsample(plane, yi + 1, xi + 1);
      const float top = v00 * (1.0f - wx) + v01 * wx;
      const float bot = v10 * (1.0f - wx) + v11 * wx;
      Bl[n * 32 + krow] = (_Float16)(top * (1.0f - wy) + bot * wy);
    }
    __syncthreads();

    v16h a = *(const v16h*)(wdp + ((size_t)(kb * 4 + wave) * 32 + lane) * 16);
    v16h f = *(const v16h*)(Bl + (lane & 15) * 32 + (lane >> 4) * 16);
    acc = __builtin_amdgcn_wmma_f32_16x16x32_f16(false, a, false, f, (short)0, acc, false, false);
    __syncthreads();
  }

  const int hi8 = (lane >> 4) * 8;
  const int n   = lane & 15;
#pragma unroll
  for (int r = 0; r < 8; ++r) {
    const int och = wave * 16 + hi8 + r;
    out[(((size_t)b * NC + och) * NH + y) * NW + x0 + n] = acc[r];
  }
}

// ---------------------------------------------------------------------------
// Workspace layout (bytes): total ~26.5 MB
//   [0)         w1p  f16  36*4*512  = 147456 B
//   [147456)    w2p  f16  18*2*512  =  36864 B
//   [184320)    wdp  f16  18*4*512  =  73728 B
//   [258048)    hmid f16  8*64*128*128 = 16777216 B
//   [17035264)  offs f32  8*18*128*128 = 9437184 B
// ---------------------------------------------------------------------------
extern "C" void kernel_launch(void* const* d_in, const int* in_sizes, int n_in,
                              void* d_out, int out_size, void* d_ws, size_t ws_size,
                              hipStream_t stream) {
  (void)in_sizes; (void)n_in; (void)out_size; (void)ws_size;
  const float* ref = (const float*)d_in[0];
  const float* tgt = (const float*)d_in[1];
  const float* w1  = (const float*)d_in[2];
  const float* b1  = (const float*)d_in[3];
  const float* w2  = (const float*)d_in[4];
  const float* b2  = (const float*)d_in[5];
  const float* wd  = (const float*)d_in[6];
  float* out = (float*)d_out;

  char* ws = (char*)d_ws;
  _Float16* w1p  = (_Float16*)(ws + 0);
  _Float16* w2p  = (_Float16*)(ws + 147456);
  _Float16* wdp  = (_Float16*)(ws + 184320);
  _Float16* hmid = (_Float16*)(ws + 258048);
  float*    offs = (float*)   (ws + 17035264);

  pack_weights<<<(73728 + 255) / 256, 256, 0, stream>>>(w1, w1p, 4, 128, 64, 73728);
  pack_weights<<<(18432 + 255) / 256, 256, 0, stream>>>(w2, w2p, 2,  64, 18, 18432);
  pack_weights<<<(36864 + 255) / 256, 256, 0, stream>>>(wd, wdp, 4,  64, 64, 36864);

  conv1_wmma <<<dim3(NW / 64, NH, NB), 128, 0, stream>>>(ref, tgt, w1p, b1, hmid);
  conv2_wmma <<<dim3(NW / 64, NH, NB), 128, 0, stream>>>(hmid, w2p, b2, offs);
  deform_wmma<<<dim3(NW / 16, NH, NB), 128, 0, stream>>>(tgt, offs, wdp, out);
}
